// KohonenMap_41042707480737
// MI455X (gfx1250) — matrix operand order repaired and checked by reading, not compiled, not run on hardware
//
#include <hip/hip_runtime.h>

typedef __attribute__((ext_vector_type(16))) _Float16 v16h;
typedef __attribute__((ext_vector_type(8)))  float    v8f;
typedef __attribute__((ext_vector_type(4)))  int      v4i;

#define N_PTS 65536
#define K_NEU 4096
#define D_DIM 512

#define ROWS_PER_WG 128      // 8 waves * 16 rows
#define COLS_PER_STAGE 32    // 2 column tiles of 16
#define NSTAGES (K_NEU / COLS_PER_STAGE)
#define LDS_COL_STRIDE 520   // halves: 512 + 8 pad (16B) -> conflict-free b128 reads
#define LDS_BUF_HALVES (COLS_PER_STAGE * LDS_COL_STRIDE)

#if __has_builtin(__builtin_amdgcn_global_load_async_to_lds_b128)
#define HAVE_ASYNC_LDS 1
#else
#define HAVE_ASYNC_LDS 0
#endif

typedef v4i __attribute__((address_space(1)))* g_v4i_p;   // global int4*
typedef v4i __attribute__((address_space(3)))* l_v4i_p;   // LDS int4*

__device__ __forceinline__ void wait_async_zero() {
#if HAVE_ASYNC_LDS
#if __has_builtin(__builtin_amdgcn_s_wait_asynccnt)
    __builtin_amdgcn_s_wait_asynccnt(0);
#else
    asm volatile("s_wait_asynccnt 0x0" ::: "memory");
#endif
#endif
}

// ------------- Kernel 1a: wh = half(w)  (one-time f32 -> f16) --------------
__global__ __launch_bounds__(256)
void cvt_kernel(const float* __restrict__ w, _Float16* __restrict__ wh) {
    const size_t i = ((size_t)blockIdx.x * 256 + threadIdx.x) * 8;
    float4 a = *(const float4*)(w + i);
    float4 b = *(const float4*)(w + i + 4);
    union { _Float16 h[8]; uint4 u; } pk;
    pk.h[0] = (_Float16)a.x; pk.h[1] = (_Float16)a.y;
    pk.h[2] = (_Float16)a.z; pk.h[3] = (_Float16)a.w;
    pk.h[4] = (_Float16)b.x; pk.h[5] = (_Float16)b.y;
    pk.h[6] = (_Float16)b.z; pk.h[7] = (_Float16)b.w;
    *(uint4*)(wh + i) = pk.u;
}

// ------------- Kernel 1b: w2[k] = sum_d (half(w[k,d]))^2 -------------------
__global__ __launch_bounds__(256)
void w2_kernel(const float* __restrict__ w, float* __restrict__ w2) {
    int wave = threadIdx.x >> 5;
    int lane = threadIdx.x & 31;
    int row  = blockIdx.x * 8 + wave;
    const float* wr = w + (size_t)row * D_DIM;
    float s = 0.0f;
    #pragma unroll
    for (int i = lane * 4; i < D_DIM; i += 32 * 4) {
        float4 v = *(const float4*)(wr + i);
        float a = (float)(_Float16)v.x;
        float b = (float)(_Float16)v.y;
        float c = (float)(_Float16)v.z;
        float d = (float)(_Float16)v.w;
        s += a * a + b * b + c * c + d * d;
    }
    #pragma unroll
    for (int m = 16; m >= 1; m >>= 1) s += __shfl_xor(s, m, 32);
    if (lane == 0) w2[row] = s;
}

// ------- Kernel 2: fused WMMA GEMM + running argmin, async 2x-buffered -----
__global__ __launch_bounds__(256)
void bmu_kernel(const float* __restrict__ x, const _Float16* __restrict__ wh,
                const float* __restrict__ w2, int* __restrict__ bmu) {
    __shared__ _Float16 lw[2 * LDS_BUF_HALVES];   // 66,560 B

    const int tid  = threadIdx.x;
    const int lane = tid & 31;
    const int l16  = lane & 15;
    const int half = lane >> 4;           // 0: rows 0-7 ; 1: rows 8-15
    const int wave = tid >> 5;
    const int rowBase = blockIdx.x * ROWS_PER_WG + wave * 16;

    // ---- Load A fragments (16 rows x 512 d) resident in VGPRs ----
    // A layout (f16 16x32): lane half h: slots 0-15 <- K = h*16 + 0..15 (contiguous)
    v16h a[16];
    {
        const float* xr = x + (size_t)(rowBase + l16) * D_DIM;
        #pragma unroll
        for (int c = 0; c < 16; ++c) {
            const int k0 = c * 32 + half * 16;
            float4 f0 = *(const float4*)(xr + k0);
            float4 f1 = *(const float4*)(xr + k0 + 4);
            float4 f2 = *(const float4*)(xr + k0 + 8);
            float4 f3 = *(const float4*)(xr + k0 + 12);
            v16h av;
            av[0]  = (_Float16)f0.x; av[1]  = (_Float16)f0.y;
            av[2]  = (_Float16)f0.z; av[3]  = (_Float16)f0.w;
            av[4]  = (_Float16)f1.x; av[5]  = (_Float16)f1.y;
            av[6]  = (_Float16)f1.z; av[7]  = (_Float16)f1.w;
            av[8]  = (_Float16)f2.x; av[9]  = (_Float16)f2.y;
            av[10] = (_Float16)f2.z; av[11] = (_Float16)f2.w;
            av[12] = (_Float16)f3.x; av[13] = (_Float16)f3.y;
            av[14] = (_Float16)f3.z; av[15] = (_Float16)f3.w;
            a[c] = av;
        }
    }

    // ---- Stage prefetch: 32 cols x 512 halves (32 KB) per stage ----
    // unit = 16 bytes; per thread: off fixed, col = (tid>>6) + 4*i  (strided)
    auto prefetch = [&](int stage, int buf) {
        const int colBase = stage * COLS_PER_STAGE;
        const int off = tid & 63;          // 16B unit within column
        const int cb  = tid >> 6;          // 0..3
        #pragma unroll
        for (int i = 0; i < 8; ++i) {
            const int c = cb + 4 * i;
            const _Float16* g = wh + (size_t)(colBase + c) * D_DIM + off * 8;
            _Float16* l = lw + buf * LDS_BUF_HALVES + c * LDS_COL_STRIDE + off * 8;
#if HAVE_ASYNC_LDS
            __builtin_amdgcn_global_load_async_to_lds_b128(
                (g_v4i_p)(uintptr_t)g,
                (l_v4i_p)(unsigned int)(uintptr_t)l, 0, 0);
#else
            uint4 v = *(const uint4*)g;
            *(uint4*)l = v;
#endif
        }
    };

    float best[8];
    int   bidx[8];
    #pragma unroll
    for (int r = 0; r < 8; ++r) { best[r] = 3.4e38f; bidx[r] = 0; }

    prefetch(0, 0);

    union BV { uint4 u[2]; v16h h; };

    #pragma unroll 1
    for (int stage = 0; stage < NSTAGES; ++stage) {
        const int buf = stage & 1;
        wait_async_zero();      // this wave's incoming-buffer writes complete
        __syncthreads();        // all waves: buf complete; prev compute done
        if (stage + 1 < NSTAGES) prefetch(stage + 1, buf ^ 1);

        const _Float16* lwbuf = lw + buf * LDS_BUF_HALVES;
        // B layout (f16 32x16): lane half h: slots 0-15 <- K = h*16 + 0..15 (contiguous)
        const _Float16* pB0 = lwbuf + l16 * LDS_COL_STRIDE + half * 16;
        const _Float16* pB1 = pB0 + 16 * LDS_COL_STRIDE;

        // ---- 2 column tiles x 16 K-chunks, B double-buffered in registers ----
        v8f acc0 = {}; v8f acc1 = {};
        BV bcur0, bcur1, bnxt0, bnxt1;
        bcur0.u[0] = *(const uint4*)(pB0);     bcur0.u[1] = *(const uint4*)(pB0 + 8);
        bcur1.u[0] = *(const uint4*)(pB1);     bcur1.u[1] = *(const uint4*)(pB1 + 8);
        #pragma unroll
        for (int c = 0; c < 16; ++c) {
            if (c < 15) {
                const _Float16* q0 = pB0 + (c + 1) * 32;
                const _Float16* q1 = pB1 + (c + 1) * 32;
                bnxt0.u[0] = *(const uint4*)(q0);  bnxt0.u[1] = *(const uint4*)(q0 + 8);
                bnxt1.u[0] = *(const uint4*)(q1);  bnxt1.u[1] = *(const uint4*)(q1 + 8);
            }
            acc0 = __builtin_amdgcn_wmma_f32_16x16x32_f16(false, a[c], false, bcur0.h,
                                                          (short)0, acc0, false, false);
            acc1 = __builtin_amdgcn_wmma_f32_16x16x32_f16(false, a[c], false, bcur1.h,
                                                          (short)0, acc1, false, false);
            bcur0 = bnxt0; bcur1 = bnxt1;   // renamed away by full unroll
        }

        // ---- Update running argmin: score = w2[n] - 2*dot ----
        const int colBase = stage * COLS_PER_STAGE;
        const int n0 = colBase + l16;
        const int n1 = colBase + 16 + l16;
        const float w2n0 = w2[n0];
        const float w2n1 = w2[n1];
        #pragma unroll
        for (int r = 0; r < 8; ++r) {
            const float s0 = w2n0 - 2.0f * acc0[r];
            if (s0 < best[r]) { best[r] = s0; bidx[r] = n0; }
            const float s1 = w2n1 - 2.0f * acc1[r];
            if (s1 < best[r]) { best[r] = s1; bidx[r] = n1; }
        }
    }

    // ---- Cross-lane reduction within each 16-lane column group ----
    #pragma unroll
    for (int r = 0; r < 8; ++r) {
        float v = best[r]; int i = bidx[r];
        #pragma unroll
        for (int m = 1; m < 16; m <<= 1) {
            float ov = __shfl_xor(v, m, 32);
            int   oi = __shfl_xor(i, m, 32);
            if (ov < v || (ov == v && oi < i)) { v = ov; i = oi; }
        }
        best[r] = v; bidx[r] = i;
    }
    if (l16 == 0) {
        const int m0 = rowBase + half * 8;   // lane 0: rows 0-7, lane 16: rows 8-15
        #pragma unroll
        for (int r = 0; r < 8; ++r) bmu[m0 + r] = bidx[r];
    }
}

// ---------------- Kernel 3: out[n] = weights[bmu[n]] -----------------------
__global__ __launch_bounds__(128)
void gather_kernel(const float* __restrict__ w, const int* __restrict__ bmu,
                   float* __restrict__ out) {
    const int row = blockIdx.x;
    const int b   = bmu[row];
    const float4* src = (const float4*)(w + (size_t)b * D_DIM);
    float4* dst = (float4*)(out + (size_t)row * D_DIM);
    dst[threadIdx.x] = src[threadIdx.x];   // 128 threads * float4 = 512 floats
}

extern "C" void kernel_launch(void* const* d_in, const int* in_sizes, int n_in,
                              void* d_out, int out_size, void* d_ws, size_t ws_size,
                              hipStream_t stream) {
    const float* x = (const float*)d_in[0];
    const float* w = (const float*)d_in[1];
    float* out = (float*)d_out;

    _Float16* wh = (_Float16*)d_ws;                                   // 4 MB
    float*    w2 = (float*)((char*)d_ws + (size_t)K_NEU * D_DIM * 2); // 16 KB
    int*     bmu = (int*)((char*)w2 + K_NEU * sizeof(float));         // 256 KB

    cvt_kernel<<<(K_NEU * D_DIM) / (256 * 8), 256, 0, stream>>>(w, wh);
    w2_kernel<<<K_NEU / 8, 256, 0, stream>>>(w, w2);
    bmu_kernel<<<N_PTS / ROWS_PER_WG, 256, 0, stream>>>(x, wh, w2, bmu);
    gather_kernel<<<N_PTS, 128, 0, stream>>>(w, bmu, out);
}
